// GPTQLinear_59012850647279
// MI455X (gfx1250) — matrix operand order repaired and checked by reading, not compiled, not run on hardware
//
#include <hip/hip_runtime.h>

// GPTQ int4 linear: out[b,s,o] = sum_k x[b,s,k] * (q[o,k]*scale[o]) + bias[o]
//   == scale[o] * (X · Qᵀ)[m,o] + bias[o],  M = B*S = 8192, N = DOUT, K = DIN.
// f16 WMMA GEMM with f32 accumulation; int4 codes are exact in f16.

#define K_DIM 4096
#define N_DIM 4096
#define M_DIM 8192

constexpr int BM = 128;        // block tile M
constexpr int BN = 128;        // block tile N
constexpr int BK = 32;         // K per LDS stage = one WMMA K-step
constexpr int LDK = BK + 8;    // padded LDS row stride (halves); 80 B, 16B-aligned rows
constexpr int WM = 64;         // wave tile M  (2 waves along M)
constexpr int WN = 32;         // wave tile N  (4 waves along N)

typedef __attribute__((ext_vector_type(16))) _Float16 v16h;
typedef __attribute__((ext_vector_type(8)))  float    v8f;

union FragU { v16h h; uint4 u[2]; };
union H16U  { _Float16 e[16]; uint4 u[2]; };

__global__ __launch_bounds__(256)
void gptq_wmma_f16(const float* __restrict__ x,
                   const int*   __restrict__ qw,
                   const float* __restrict__ scale,
                   const float* __restrict__ bias,
                   float*       __restrict__ out)
{
    __shared__ __align__(16) _Float16 sA[2][BM * LDK];
    __shared__ __align__(16) _Float16 sB[2][BN * LDK];

    const int tid   = threadIdx.x;
    const int lane  = tid & 31;
    const int wid   = tid >> 5;
    const int waveM = (wid >> 2) * WM;     // 0 / 64
    const int waveN = (wid & 3)  * WN;     // 0 / 32 / 64 / 96
    const int blockN = blockIdx.x * BN;
    const int blockM = blockIdx.y * BM;

    // ---- cooperative tile loaders: 256 threads, each owns half a row (16 elems)
    const int ldRow = tid >> 1;            // 0..127
    const int ldKof = (tid & 1) * 16;      // 0 or 16

    const float* gA = x  + (size_t)(blockM + ldRow) * K_DIM + ldKof;
    const int*   gB = qw + (size_t)(blockN + ldRow) * K_DIM + ldKof;

    float4 ra[4];
    int4   rb[4];

    auto gload = [&](int kt) {
        const float4* pa = (const float4*)(gA + (size_t)kt * BK);
        const int4*   pb = (const int4*)  (gB + (size_t)kt * BK);
        #pragma unroll
        for (int i = 0; i < 4; ++i) { ra[i] = pa[i]; rb[i] = pb[i]; }
    };

    auto sstore = [&](int buf) {
        H16U ha, hb;
        #pragma unroll
        for (int i = 0; i < 4; ++i) {
            ha.e[4*i+0] = (_Float16)ra[i].x;  ha.e[4*i+1] = (_Float16)ra[i].y;
            ha.e[4*i+2] = (_Float16)ra[i].z;  ha.e[4*i+3] = (_Float16)ra[i].w;
            hb.e[4*i+0] = (_Float16)(float)rb[i].x;  hb.e[4*i+1] = (_Float16)(float)rb[i].y;
            hb.e[4*i+2] = (_Float16)(float)rb[i].z;  hb.e[4*i+3] = (_Float16)(float)rb[i].w;
        }
        _Float16* pa = &sA[buf][ldRow * LDK + ldKof];
        _Float16* pb = &sB[buf][ldRow * LDK + ldKof];
        *(uint4*)pa        = ha.u[0];
        *(uint4*)(pa + 8)  = ha.u[1];
        *(uint4*)pb        = hb.u[0];
        *(uint4*)(pb + 8)  = hb.u[1];
    };

    // ---- fragment loads per ISA 7.12.2 (wave32)
    // A 16x32 f16: lane r=lane&15; VGPR0-3 = K[(lane>>4)*8 ..+8), VGPR4-7 = +16.
    auto loadA = [&](int buf, int mi) -> v16h {
        const int r  = waveM + mi * 16 + (lane & 15);
        const int kh = (lane >> 4) * 8;
        const _Float16* p = &sA[buf][r * LDK + kh];
        FragU f;
        f.u[0] = *(const uint4*)p;
        f.u[1] = *(const uint4*)(p + 16);
        return f.h;
    };
    // B 32x16 f16: lane n=lane&15; VGPR0-3 = K[(lane>>4)*8 ..+8), VGPR4-7 = +16.
    auto loadB = [&](int buf, int ni) -> v16h {
        const int r  = waveN + ni * 16 + (lane & 15);
        const int ks = (lane >> 4) * 8;
        const _Float16* p = &sB[buf][r * LDK + ks];
        FragU f;
        f.u[0] = *(const uint4*)p;
        f.u[1] = *(const uint4*)(p + 16);
        return f.h;
    };

    v8f acc[4][2];
    #pragma unroll
    for (int mi = 0; mi < 4; ++mi)
        #pragma unroll
        for (int ni = 0; ni < 2; ++ni)
            acc[mi][ni] = (v8f){0.f, 0.f, 0.f, 0.f, 0.f, 0.f, 0.f, 0.f};

    // ---- software-pipelined main loop, double-buffered LDS
    gload(0);
    sstore(0);
    __syncthreads();

    const int NK = K_DIM / BK;             // 128
    for (int kt = 0; kt < NK; ++kt) {
        const int cur = kt & 1;
        if (kt + 1 < NK) gload(kt + 1);    // prefetch next global tile into regs

        v16h a[4], b[2];
        #pragma unroll
        for (int mi = 0; mi < 4; ++mi) a[mi] = loadA(cur, mi);
        #pragma unroll
        for (int ni = 0; ni < 2; ++ni) b[ni] = loadB(cur, ni);

        #pragma unroll
        for (int mi = 0; mi < 4; ++mi)
            #pragma unroll
            for (int ni = 0; ni < 2; ++ni)
                acc[mi][ni] = __builtin_amdgcn_wmma_f32_16x16x32_f16(
                    /*neg_a=*/false, a[mi], /*neg_b=*/false, b[ni],
                    /*c_mod=*/(short)0, acc[mi][ni],
                    /*reuse_a=*/false, /*reuse_b=*/false);

        if (kt + 1 < NK) sstore(cur ^ 1);  // fill the other buffer
        __syncthreads();                   // reads(cur) done; writes(cur^1) visible
    }

    // ---- epilogue: scale[n]*acc + bias[n]
    // C layout: lane L, vgpr r -> m = r + 8*(L>>4), n = L&15.
    #pragma unroll
    for (int ni = 0; ni < 2; ++ni) {
        const int n = blockN + waveN + ni * 16 + (lane & 15);
        const float s  = scale[n];
        const float bz = bias[n];
        #pragma unroll
        for (int mi = 0; mi < 4; ++mi) {
            const int m0 = blockM + waveM + mi * 16 + ((lane >> 4) << 3);
            #pragma unroll
            for (int r = 0; r < 8; ++r)
                out[(size_t)(m0 + r) * N_DIM + n] = acc[mi][ni][r] * s + bz;
        }
    }
}

extern "C" void kernel_launch(void* const* d_in, const int* in_sizes, int n_in,
                              void* d_out, int out_size, void* d_ws, size_t ws_size,
                              hipStream_t stream)
{
    const float* x     = (const float*)d_in[0];   // [4,2048,4096] f32
    const int*   qw    = (const int*)  d_in[1];   // [4096,4096] i32 codes in [-8,7]
    const float* scale = (const float*)d_in[2];   // [4096] f32
    const float* bias  = (const float*)d_in[3];   // [4096] f32
    float*       out   = (float*)d_out;           // [4,2048,4096] f32

    dim3 grid(N_DIM / BN, M_DIM / BM);            // (32, 64): M-adjacent blocks share W panel in L2
    dim3 block(256);                              // 8 wave32 waves
    gptq_wmma_f16<<<grid, block, 0, stream>>>(x, qw, scale, bias, out);
}
